// CrossOrganismAttention_58317065945713
// MI455X (gfx1250) — matrix operand order repaired and checked by reading, not compiled, not run on hardware
//
#include <hip/hip_runtime.h>
#include <hip/hip_bf16.h>
#include <math.h>

// CrossOrganismAttention fused block for gfx1250 (MI455X).
// B=65536, S=3, D=256, H=4, DH=64.

#define BATCH   65536
#define SEQ     3
#define DIM     256
#define NHEAD   4
#define DHEAD   64
#define ROWS_WG 16
#define TOK_WG  (ROWS_WG * SEQ)      // 48 tokens per workgroup (3 M-tiles of 16)
#define NWAVES  8
#define THREADS 256

typedef __attribute__((ext_vector_type(16))) __bf16 v16bf;
typedef __attribute__((ext_vector_type(8)))  __bf16 v8bf;
typedef __attribute__((ext_vector_type(8)))  float  v8f;

// ---------------- LDS layout (bytes) ----------------
// tokX : fp32 [48][256]   residual / post-LN activations      49152
// qkv  : fp32 [48][768]   q|k|v per token                    147456
// abf  : bf16 [48][256]   WMMA A operand staging              24576
// hbf  : bf16 [48][512]   FFN hidden (A for FFN2)             49152
// pmat : fp32 [16][4][3][3] softmax probabilities              2304
constexpr int TOKX_OFF   = 0;
constexpr int QKV_OFF    = TOKX_OFF + TOK_WG * DIM * 4;            // 49152
constexpr int ABF_OFF    = QKV_OFF  + TOK_WG * 3 * DIM * 4;        // 196608
constexpr int HBF_OFF    = ABF_OFF  + TOK_WG * DIM * 2;            // 221184
constexpr int P_OFF      = HBF_OFF  + TOK_WG * 2 * DIM * 2;        // 270336
constexpr int SMEM_BYTES = P_OFF + ROWS_WG * NHEAD * SEQ * SEQ * 4; // 272640 (<320KB WGP LDS)

__device__ __forceinline__ float gelu_exact(float x) {
    return 0.5f * x * (1.0f + erff(x * 0.70710678118654752440f));
}

// ---- one wave computes a 48x16 output slab: 3 WMMA M-tiles, shared B tile ----
// A layout (16x32 bf16, ISA 7.12.2): lane(l2,hi) -> row M=l2,
//   chunk0 = K[kb + hi*8 .. +7], chunk1 = K[kb + 16 + hi*8 .. +7]
// B layout (32x16 bf16): lane(l2,hi) -> col N=l2 == weight row (x@W^T),
//   chunk0 = K[kb + hi*8 .. +7], chunk1 = K[kb + 16 + hi*8 .. +7] (same as A).
template<int KTILES, int LDA, int LDB>
__device__ __forceinline__ void wmma_tile3(const __bf16* __restrict__ A,
                                           const __bf16* __restrict__ Bw,
                                           int nbase, v8f (&acc)[3])
{
    const int lane = threadIdx.x & 31;
    const int l2   = lane & 15;
    const int hi   = lane >> 4;
    const __bf16* brow = Bw + (size_t)(nbase + l2) * LDB + hi * 8;
#pragma unroll
    for (int kt = 0; kt < KTILES; ++kt) {
        const int kb = kt * 32;
        v8bf b0 = *(const v8bf*)(brow + kb);
        v8bf b1 = *(const v8bf*)(brow + kb + 16);
        v16bf bb = __builtin_shufflevector(b0, b1, 0,1,2,3,4,5,6,7,8,9,10,11,12,13,14,15);
#pragma unroll
        for (int mt = 0; mt < 3; ++mt) {
            const __bf16* arow = A + (size_t)(mt * 16 + l2) * LDA + hi * 8;
            v8bf a0 = *(const v8bf*)(arow + kb);
            v8bf a1 = *(const v8bf*)(arow + kb + 16);
            v16bf aa = __builtin_shufflevector(a0, a1, 0,1,2,3,4,5,6,7,8,9,10,11,12,13,14,15);
            acc[mt] = __builtin_amdgcn_wmma_f32_16x16x32_bf16(
                false, aa, false, bb, (short)0, acc[mt], false, false);
        }
    }
}

// LayerNorm over D=256 per token; 8 waves each own tokens wave, wave+8, ...
// 32 lanes x 8 elements; wave32 shuffle reduction.
__device__ __forceinline__ void layernorm_rows(float* tokX, __bf16* abf,
                                               const float* __restrict__ g,
                                               const float* __restrict__ b)
{
    const int wave = threadIdx.x >> 5;
    const int lane = threadIdx.x & 31;
    for (int m = wave; m < TOK_WG; m += NWAVES) {
        float* row = tokX + m * DIM;
        float x[8];
        float s = 0.f, s2 = 0.f;
#pragma unroll
        for (int i = 0; i < 8; ++i) {
            x[i] = row[lane * 8 + i];
            s  += x[i];
            s2 += x[i] * x[i];
        }
#pragma unroll
        for (int o = 16; o > 0; o >>= 1) {
            s  += __shfl_xor(s,  o, 32);
            s2 += __shfl_xor(s2, o, 32);
        }
        const float mean = s * (1.0f / DIM);
        const float var  = s2 * (1.0f / DIM) - mean * mean;
        const float inv  = rsqrtf(var + 1e-5f);
#pragma unroll
        for (int i = 0; i < 8; ++i) {
            const int d = lane * 8 + i;
            const float y = (x[i] - mean) * inv * g[d] + b[d];
            row[d] = y;
            if (abf) abf[m * DIM + d] = (__bf16)y;
        }
    }
}

__global__ void __launch_bounds__(THREADS, 1)
cvt_f32_to_bf16(const float* __restrict__ src, unsigned short* __restrict__ dst, int n)
{
    int i = blockIdx.x * blockDim.x + threadIdx.x;
    if (i < n) {
        __bf16 h = (__bf16)src[i];
        dst[i] = __builtin_bit_cast(unsigned short, h);
    }
}

__global__ void __launch_bounds__(THREADS, 1)
cross_organism_fused(const float* __restrict__ org,        // (B,S,D)
                     const float* __restrict__ anom,       // (B,S)
                     const unsigned char* __restrict__ msk,// (B,S) bool
                     const float* __restrict__ sp_w1,      // (D)
                     const float* __restrict__ sp_b1,      // (D)
                     const float* __restrict__ sp_b2,      // (D)
                     const float* __restrict__ species,    // (S,D)
                     const float* __restrict__ in_b,       // (3D)
                     const float* __restrict__ out_b,      // (D)
                     const float* __restrict__ ln1_g, const float* __restrict__ ln1_b,
                     const float* __restrict__ ln2_g, const float* __restrict__ ln2_b,
                     const float* __restrict__ ffn_b1,     // (2D)
                     const float* __restrict__ ffn_b2,     // (D)
                     const unsigned short* __restrict__ w_sp2, // (D,D)  bf16
                     const unsigned short* __restrict__ w_in,  // (3D,D) bf16
                     const unsigned short* __restrict__ w_out, // (D,D)  bf16
                     const unsigned short* __restrict__ w_f1,  // (2D,D) bf16
                     const unsigned short* __restrict__ w_f2,  // (D,2D) bf16
                     float* __restrict__ out)              // (B,D)
{
    extern __shared__ char smem[];
    float*  tokX = (float*)(smem + TOKX_OFF);
    float*  qkv  = (float*)(smem + QKV_OFF);
    __bf16* abf  = (__bf16*)(smem + ABF_OFF);
    __bf16* hbf  = (__bf16*)(smem + HBF_OFF);
    float*  pmat = (float*)(smem + P_OFF);

    const int tid  = threadIdx.x;
    const int wave = tid >> 5;
    const int lane = tid & 31;
    const int l2   = lane & 15;
    const int hi   = lane >> 4;
    const int wg   = blockIdx.x;                // 4096 groups of 16 batch rows

    // ---------------- Stage 0: h = gelu(score * sp_w1 + sp_b1) (bf16 A) -------
    for (int e = tid; e < TOK_WG * DIM; e += THREADS) {
        const int m = e >> 8, d = e & 255;
        const size_t g = (size_t)wg * TOK_WG + m;   // global token = b*3+s
        const float h = anom[g] * sp_w1[d] + sp_b1[d];
        abf[e] = (__bf16)gelu_exact(h);
    }
    __syncthreads();

    // ---------------- Stage 1: tokens = org + species + h @ sp_w2^T + sp_b2 ---
    for (int nt = wave; nt < DIM / 16; nt += NWAVES) {
        const int nbase = nt * 16;
        v8f acc[3] = {};
        wmma_tile3<8, DIM, DIM>(abf, (const __bf16*)w_sp2, nbase, acc);
        const int n = nbase + l2;
        const float bias = sp_b2[n];
#pragma unroll
        for (int mt = 0; mt < 3; ++mt)
#pragma unroll
            for (int r = 0; r < 8; ++r) {
                const int m = mt * 16 + r + 8 * hi;
                const int s = m % SEQ;
                const size_t g = (size_t)wg * TOK_WG + m;
                tokX[m * DIM + n] = acc[mt][r] + bias + org[g * DIM + n] + species[s * DIM + n];
            }
    }
    __syncthreads();

    // tokens -> bf16 A
    for (int e = tid; e < TOK_WG * DIM; e += THREADS) abf[e] = (__bf16)tokX[e];
    __syncthreads();

    // ---------------- Stage 2: qkv = tokens @ in_w^T + in_b -------------------
    for (int nt = wave; nt < (3 * DIM) / 16; nt += NWAVES) {
        const int nbase = nt * 16;
        v8f acc[3] = {};
        wmma_tile3<8, DIM, DIM>(abf, (const __bf16*)w_in, nbase, acc);
        const int n = nbase + l2;
        const float bias = in_b[n];
#pragma unroll
        for (int mt = 0; mt < 3; ++mt)
#pragma unroll
            for (int r = 0; r < 8; ++r) {
                const int m = mt * 16 + r + 8 * hi;
                qkv[m * (3 * DIM) + n] = acc[mt][r] + bias;
            }
    }
    __syncthreads();

    // ---------------- Stage 3: masked softmax attention (S=3) ------------------
    if (tid < ROWS_WG * NHEAD) {                 // one (row, head) per thread
        const int r = tid >> 2, h = tid & 3;
        const size_t brow = (size_t)wg * ROWS_WG + r;
        float sc[SEQ][SEQ];
#pragma unroll
        for (int q = 0; q < SEQ; ++q)
#pragma unroll
            for (int k = 0; k < SEQ; ++k) {
                const float* qv = qkv + (r * SEQ + q) * (3 * DIM) + h * DHEAD;
                const float* kv = qkv + (r * SEQ + k) * (3 * DIM) + DIM + h * DHEAD;
                float d = 0.f;
                for (int i = 0; i < DHEAD; ++i) d += qv[i] * kv[i];
                sc[q][k] = msk[brow * SEQ + k] ? -1e9f : d * 0.125f; // 1/sqrt(64)
            }
#pragma unroll
        for (int q = 0; q < SEQ; ++q) {
            const float mx = fmaxf(sc[q][0], fmaxf(sc[q][1], sc[q][2]));
            float e0 = __expf(sc[q][0] - mx);
            float e1 = __expf(sc[q][1] - mx);
            float e2 = __expf(sc[q][2] - mx);
            const float inv = 1.0f / (e0 + e1 + e2);
            float* p = pmat + ((r * NHEAD + h) * SEQ + q) * SEQ;
            p[0] = e0 * inv; p[1] = e1 * inv; p[2] = e2 * inv;
        }
    }
    __syncthreads();

    // o = p @ v  -> bf16 A for out-projection
    for (int e = tid; e < TOK_WG * DIM; e += THREADS) {
        const int m = e >> 8, c = e & 255;
        const int r = m / SEQ, q = m % SEQ, h = c >> 6;
        const float* p = pmat + ((r * NHEAD + h) * SEQ + q) * SEQ;
        float o = 0.f;
#pragma unroll
        for (int k = 0; k < SEQ; ++k)
            o += p[k] * qkv[(r * SEQ + k) * (3 * DIM) + 2 * DIM + c];
        abf[e] = (__bf16)o;
    }
    __syncthreads();

    // ---------------- Stage 4: x = tokens + o @ out_w^T + out_b ----------------
    for (int nt = wave; nt < DIM / 16; nt += NWAVES) {
        const int nbase = nt * 16;
        v8f acc[3] = {};
        wmma_tile3<8, DIM, DIM>(abf, (const __bf16*)w_out, nbase, acc);
        const int n = nbase + l2;
        const float bias = out_b[n];
#pragma unroll
        for (int mt = 0; mt < 3; ++mt)
#pragma unroll
            for (int r = 0; r < 8; ++r) {
                const int m = mt * 16 + r + 8 * hi;
                tokX[m * DIM + n] += acc[mt][r] + bias;   // residual
            }
    }
    __syncthreads();

    layernorm_rows(tokX, abf, ln1_g, ln1_b);   // post-LN kept fp32 + bf16 A
    __syncthreads();

    // ---------------- Stage 5: FFN1 = gelu(x @ w1^T + b1) (N=512) -------------
    for (int nt = wave; nt < (2 * DIM) / 16; nt += NWAVES) {
        const int nbase = nt * 16;
        v8f acc[3] = {};
        wmma_tile3<8, DIM, DIM>(abf, (const __bf16*)w_f1, nbase, acc);
        const int n = nbase + l2;
        const float bias = ffn_b1[n];
#pragma unroll
        for (int mt = 0; mt < 3; ++mt)
#pragma unroll
            for (int r = 0; r < 8; ++r) {
                const int m = mt * 16 + r + 8 * hi;
                hbf[m * (2 * DIM) + n] = (__bf16)gelu_exact(acc[mt][r] + bias);
            }
    }
    __syncthreads();

    // ---------------- Stage 6: x = x + h @ w2^T + b2 (K=512) ------------------
    for (int nt = wave; nt < DIM / 16; nt += NWAVES) {
        const int nbase = nt * 16;
        v8f acc[3] = {};
        wmma_tile3<16, 2 * DIM, 2 * DIM>(hbf, (const __bf16*)w_f2, nbase, acc);
        const int n = nbase + l2;
        const float bias = ffn_b2[n];
#pragma unroll
        for (int mt = 0; mt < 3; ++mt)
#pragma unroll
            for (int r = 0; r < 8; ++r) {
                const int m = mt * 16 + r + 8 * hi;
                tokX[m * DIM + n] += acc[mt][r] + bias;   // residual
            }
    }
    __syncthreads();

    layernorm_rows(tokX, nullptr, ln2_g, ln2_b);
    __syncthreads();

    // ---------------- Stage 7: masked mean pool over S -------------------------
    for (int e = tid; e < ROWS_WG * DIM; e += THREADS) {
        const int r = e >> 8, d = e & 255;
        const size_t brow = (size_t)wg * ROWS_WG + r;
        float sum = 0.f, cnt = 0.f;
#pragma unroll
        for (int s = 0; s < SEQ; ++s) {
            if (!msk[brow * SEQ + s]) {
                sum += tokX[(r * SEQ + s) * DIM + d];
                cnt += 1.f;
            }
        }
        out[brow * DIM + d] = sum / fmaxf(cnt, 1.f);
    }
}

extern "C" void kernel_launch(void* const* d_in, const int* in_sizes, int n_in,
                              void* d_out, int out_size, void* d_ws, size_t ws_size,
                              hipStream_t stream) {
    const float* org     = (const float*)d_in[0];
    const float* anom    = (const float*)d_in[1];
    const unsigned char* msk = (const unsigned char*)d_in[2];
    const float* sp_w1   = (const float*)d_in[3];
    const float* sp_b1   = (const float*)d_in[4];
    const float* sp_w2   = (const float*)d_in[5];
    const float* sp_b2   = (const float*)d_in[6];
    const float* species = (const float*)d_in[7];
    const float* in_w    = (const float*)d_in[8];
    const float* in_b    = (const float*)d_in[9];
    const float* out_w   = (const float*)d_in[10];
    const float* out_b   = (const float*)d_in[11];
    const float* ln1_g   = (const float*)d_in[12];
    const float* ln1_b   = (const float*)d_in[13];
    const float* ln2_g   = (const float*)d_in[14];
    const float* ln2_b   = (const float*)d_in[15];
    const float* ffn_w1  = (const float*)d_in[16];
    const float* ffn_b1  = (const float*)d_in[17];
    const float* ffn_w2  = (const float*)d_in[18];
    const float* ffn_b2  = (const float*)d_in[19];

    // bf16 weight staging in workspace (re-generated every call: deterministic)
    unsigned short* ws = (unsigned short*)d_ws;
    unsigned short* w_sp2 = ws;                 // 256*256   = 65536
    unsigned short* w_in  = ws + 65536;         // 768*256   = 196608
    unsigned short* w_out = ws + 262144;        // 256*256   = 65536
    unsigned short* w_f1  = ws + 327680;        // 512*256   = 131072
    unsigned short* w_f2  = ws + 458752;        // 256*512   = 131072

    cvt_f32_to_bf16<<<(65536  + THREADS - 1) / THREADS, THREADS, 0, stream>>>(sp_w2,  w_sp2, 65536);
    cvt_f32_to_bf16<<<(196608 + THREADS - 1) / THREADS, THREADS, 0, stream>>>(in_w,   w_in,  196608);
    cvt_f32_to_bf16<<<(65536  + THREADS - 1) / THREADS, THREADS, 0, stream>>>(out_w,  w_out, 65536);
    cvt_f32_to_bf16<<<(131072 + THREADS - 1) / THREADS, THREADS, 0, stream>>>(ffn_w1, w_f1,  131072);
    cvt_f32_to_bf16<<<(131072 + THREADS - 1) / THREADS, THREADS, 0, stream>>>(ffn_w2, w_f2,  131072);

    const int grid = BATCH / ROWS_WG;           // 4096 workgroups
    cross_organism_fused<<<grid, THREADS, SMEM_BYTES, stream>>>(
        org, anom, msk, sp_w1, sp_b1, sp_b2, species, in_b, out_b,
        ln1_g, ln1_b, ln2_g, ln2_b, ffn_b1, ffn_b2,
        w_sp2, w_in, w_out, w_f1, w_f2, (float*)d_out);
}
